// SpatialAttn2DBlock_52140902973460
// MI455X (gfx1250) — matrix operand order repaired and checked by reading, not compiled, not run on hardware
//
#include <hip/hip_runtime.h>
#include <hip/hip_bf16.h>

// ---------------------------------------------------------------------------
// SpatialAttn2D block for MI455X (gfx1250): f16 WMMA GEMMs + flash attention
// B=8, C=512, H=W=32 (HW=1024), NH=8, DH=64.  M = B*HW = 8192 tokens.
// ---------------------------------------------------------------------------

#define Bc   8
#define Cc   512
#define HWc  1024
#define NHc  8
#define DHc  64
#define Mc   (Bc * HWc)   // 8192

typedef __attribute__((ext_vector_type(16))) _Float16 v16h;
typedef __attribute__((ext_vector_type(8)))  _Float16 v8h;
typedef __attribute__((ext_vector_type(8)))  float    v8f;

// ---- gfx1250 async global->LDS copy (ASYNCcnt path), with safe fallback ----
#if defined(__AMDGCN__) && __has_builtin(__builtin_amdgcn_global_load_async_to_lds_b128)
#define HAVE_ASYNC_LDS 1
typedef __attribute__((ext_vector_type(4))) int v4i_;
typedef __attribute__((address_space(1))) v4i_ gv4i;   // global (AS1) int4
typedef __attribute__((address_space(3))) v4i_ lv4i;   // LDS    (AS3) int4
#if __has_builtin(__builtin_amdgcn_s_wait_asynccnt)
#define ASYNC_WAIT() __builtin_amdgcn_s_wait_asynccnt(0)
#else
#define ASYNC_WAIT() asm volatile("s_wait_asynccnt 0x0" ::: "memory")
#endif
#else
#define HAVE_ASYNC_LDS 0
#define ASYNC_WAIT()
#endif

__device__ __forceinline__ v8f zero8() {
  v8f z;
#pragma unroll
  for (int i = 0; i < 8; ++i) z[i] = 0.0f;
  return z;
}

// assemble a v16h A/B fragment from two contiguous 16-byte chunks
__device__ __forceinline__ v16h ld2(const _Float16* p0, const _Float16* p1) {
  v8h a = *(const v8h*)p0;
  v8h b = *(const v8h*)p1;
  v16h r;
#pragma unroll
  for (int i = 0; i < 8; ++i) { r[i] = a[i]; r[i + 8] = b[i]; }
  return r;
}

__device__ __forceinline__ v8f wmma16(v16h a, v16h b, v8f c) {
  // D = A(16x32 f16) x B(32x16 f16) + C(16x16 f32)
  return __builtin_amdgcn_wmma_f32_16x16x32_f16(false, a, false, b,
                                                (short)0, c, false, false);
}

// ---------------------------------------------------------------------------
// weight transpose+convert: src f32 [N][K] -> dst f16 [K][N]
// ---------------------------------------------------------------------------
__global__ void k_wt(const float* __restrict__ src, _Float16* __restrict__ dst,
                     int N, int K) {
  int idx = blockIdx.x * 256 + threadIdx.x;
  if (idx >= N * K) return;
  int kk = idx / N, nn = idx % N;
  dst[idx] = (_Float16)src[(size_t)nn * K + kk];
}

// tokens: tkn[b][p][c] = x[b][c][p] + pos[p][c]   (f16)
__global__ void k_tokens(const float* __restrict__ x,
                         const float* __restrict__ pos,
                         _Float16* __restrict__ tkn) {
  int idx = blockIdx.x * 256 + threadIdx.x;       // B*HW*C
  int c = idx & (Cc - 1), p = (idx >> 9) & (HWc - 1), b = idx >> 19;
  tkn[idx] = (_Float16)(x[((size_t)b * Cc + c) * HWc + p] +
                        pos[(size_t)p * Cc + c]);
}

// ---------------------------------------------------------------------------
// Tiled WMMA GEMM:  D[M,N] = A[M,K] * Bm[K,N]  (f16 in, f32 acc)
// block tile 64x64, 128 threads = 4 waves, each wave: 16(M) x 64(N)
// Double-buffered LDS; A tile staged with async global->LDS when available.
// MODE 0: qkv scatter (q/k token-major, v d-major), MODE 1: proj(+bias)->f32,
// MODE 2: pointwise conv (+bias) + residual -> d_out
// ---------------------------------------------------------------------------
template <int MODE>
__global__ __launch_bounds__(128) void k_gemm(
    const _Float16* __restrict__ A, const _Float16* __restrict__ Bm,
    int N, int K, const float* __restrict__ bias,
    _Float16* __restrict__ qb, _Float16* __restrict__ kb,
    _Float16* __restrict__ vb, float* __restrict__ yout,
    const float* __restrict__ rbuf, float* __restrict__ outp) {
  __shared__ __align__(32) _Float16 As[2][64][32];  // A tile, row-major
  __shared__ __align__(32) _Float16 Bs[2][64][32];  // B tile, transposed [n][k]

  const int tid  = threadIdx.x;
  const int w    = tid >> 5, lane = tid & 31;
  const int lh   = lane & 15;
  const int aSel = (lane < 16) ? 0 : 8;    // A-frag K base
  const int bSel = (lane < 16) ? 0 : 8;    // B-frag K base
  const int bm = blockIdx.y * 64, bn = blockIdx.x * 64;

  v8f acc[4];
#pragma unroll
  for (int nt = 0; nt < 4; ++nt) acc[nt] = zero8();

  const int arow = tid >> 1, aoff = (tid & 1) * 16;
  const int brow = tid >> 2, boff = (tid & 3) * 16;

  // stage tile at k0 into LDS buffer `buf`
  auto stage = [&](int k0, int buf) {
    const _Float16* gA = A + (size_t)(bm + arow) * K + k0 + aoff;
    const _Float16* gB = Bm + (size_t)(k0 + brow) * N + bn + boff;
#if HAVE_ASYNC_LDS
    __builtin_amdgcn_global_load_async_to_lds_b128(
        (gv4i*)gA, (lv4i*)&As[buf][arow][aoff], 0, 0);
    __builtin_amdgcn_global_load_async_to_lds_b128(
        (gv4i*)(gA + 8), (lv4i*)&As[buf][arow][aoff + 8], 0, 0);
#else
    v8h a0 = *(const v8h*)gA;
    v8h a1 = *(const v8h*)(gA + 8);
    *(v8h*)&As[buf][arow][aoff]     = a0;
    *(v8h*)&As[buf][arow][aoff + 8] = a1;
#endif
    v8h b0 = *(const v8h*)gB;
    v8h b1 = *(const v8h*)(gB + 8);
#pragma unroll
    for (int i = 0; i < 8; ++i) {
      Bs[buf][boff + i][brow]     = b0[i];
      Bs[buf][boff + 8 + i][brow] = b1[i];
    }
  };

  const int nk = K >> 5;
  stage(0, 0);
  for (int ki = 0; ki < nk; ++ki) {
    const int cur = ki & 1;
    ASYNC_WAIT();        // this wave's async tile(s) have landed in LDS
    __syncthreads();     // all waves staged tile `cur`; prior reads retired
    if (ki + 1 < nk) stage((ki + 1) << 5, cur ^ 1);  // overlap next stage
    // A frag: lane row = w*16+lh, halves K = aSel+{0..7} and aSel+{16..23}
    v16h af = ld2(&As[cur][w * 16 + lh][aSel], &As[cur][w * 16 + lh][aSel + 16]);
#pragma unroll
    for (int nt = 0; nt < 4; ++nt) {
      // B frag: col n = nt*16+lh, halves K = bSel+{0..7} and bSel+{16..23}
      v16h bf = ld2(&Bs[cur][nt * 16 + lh][bSel], &Bs[cur][nt * 16 + lh][bSel + 16]);
      acc[nt] = wmma16(af, bf, acc[nt]);
    }
  }

#pragma unroll
  for (int nt = 0; nt < 4; ++nt) {
#pragma unroll
    for (int j = 0; j < 8; ++j) {
      int row = (lane < 16) ? j : (j + 8);
      int m = bm + w * 16 + row;
      int n = bn + nt * 16 + lh;
      float val = acc[nt][j];
      if (MODE == 0) {
        int which = n >> 9, rem = n & 511, hh = rem >> 6, dd = rem & 63;
        int b2 = m >> 10, pp = m & 1023;
        _Float16 hv = (_Float16)val;
        if (which == 0)
          qb[(((size_t)b2 * NHc + hh) * HWc + pp) * DHc + dd] = hv;
        else if (which == 1)
          kb[(((size_t)b2 * NHc + hh) * HWc + pp) * DHc + dd] = hv;
        else
          vb[(((size_t)b2 * NHc + hh) * DHc + dd) * HWc + pp] = hv;  // d-major
      } else if (MODE == 1) {
        yout[(size_t)m * Cc + n] = val + bias[n];
      } else {
        int b2 = m >> 10, pp = m & 1023;
        size_t o = ((size_t)b2 * Cc + n) * HWc + pp;
        outp[o] = rbuf[o] + val + bias[n];
      }
    }
  }
}

// ---------------------------------------------------------------------------
// token-axis L2 normalization of q or k, in place (per (b,h,d) column)
// ---------------------------------------------------------------------------
__global__ __launch_bounds__(256) void k_normqk(_Float16* __restrict__ q,
                                                _Float16* __restrict__ k) {
  __shared__ float red[256];
  int which = blockIdx.x & 1;
  int head = blockIdx.x >> 1;                // b*NH + h
  _Float16* p = (which ? k : q) + (size_t)head * HWc * DHc;
  int tid = threadIdx.x, d = tid & 63, seg = tid >> 6;
  float ss = 0.f;
  for (int t = seg * 256; t < seg * 256 + 256; ++t) {
    float v = (float)p[(size_t)t * DHc + d];
    ss += v * v;
  }
  red[tid] = ss;
  __syncthreads();
  if (seg == 0) {
    float tot = red[d] + red[64 + d] + red[128 + d] + red[192 + d];
    red[d] = 1.0f / fmaxf(sqrtf(tot), 1e-12f);
  }
  __syncthreads();
  float sc = red[d];
  for (int t = seg * 256; t < seg * 256 + 256; ++t) {
    size_t o = (size_t)t * DHc + d;
    p[o] = (_Float16)((float)p[o] * sc);
  }
}

// ---------------------------------------------------------------------------
// Flash attention: one wave = 16 query rows, online softmax, all WMMA.
// grid = B*NH*(HW/64), 128 threads (4 waves).
// ---------------------------------------------------------------------------
__global__ __launch_bounds__(128) void k_attn(
    const _Float16* __restrict__ q, const _Float16* __restrict__ kk,
    const _Float16* __restrict__ v, const float* __restrict__ temp,
    _Float16* __restrict__ obuf) {
  __shared__ __align__(32) _Float16 Psh[4][16][32];  // per-wave P bounce
  int tid = threadIdx.x, w = tid >> 5, lane = tid & 31, lh = lane & 15;
  int aSel = (lane < 16) ? 0 : 8, bSel = (lane < 16) ? 0 : 8;
  int idx = blockIdx.x;
  int s = idx & 15, h = (idx >> 4) & 7, b = idx >> 7;
  int q0 = s * 64 + w * 16;

  const _Float16* Q  = q  + (size_t)(b * NHc + h) * HWc * DHc;
  const _Float16* Kp = kk + (size_t)(b * NHc + h) * HWc * DHc;
  const _Float16* Vp = v  + (size_t)(b * NHc + h) * DHc * HWc;  // [d][tok]
  float tf = temp[h];

  int qrow = q0 + lh;
  v16h qf0 = ld2(Q + (size_t)qrow * DHc + aSel,
                 Q + (size_t)qrow * DHc + aSel + 16);
  v16h qf1 = ld2(Q + (size_t)qrow * DHc + 32 + aSel,
                 Q + (size_t)qrow * DHc + 32 + aSel + 16);

  float mi[8], li[8];
  v8f acc[4];
#pragma unroll
  for (int j = 0; j < 8; ++j) { mi[j] = -1e30f; li[j] = 0.0f; }
#pragma unroll
  for (int nt = 0; nt < 4; ++nt) acc[nt] = zero8();

  for (int t0 = 0; t0 < HWc; t0 += 32) {
    if (t0 + 32 < HWc) {  // prefetch next K/V slab (global_prefetch_b8)
      __builtin_prefetch(Kp + (size_t)(t0 + 32 + lh) * DHc, 0, 1);
      __builtin_prefetch(Vp + (size_t)lh * HWc + t0 + 32, 0, 1);
    }
    v8f s0 = zero8(), s1 = zero8();
    {  // S = Q K^T over d=64 (two k-steps), two 16-col tiles
      v16h kb0 = ld2(Kp + (size_t)(t0 + lh) * DHc + bSel,
                     Kp + (size_t)(t0 + lh) * DHc + bSel + 16);
      s0 = wmma16(qf0, kb0, s0);
      v16h kb1 = ld2(Kp + (size_t)(t0 + 16 + lh) * DHc + bSel,
                     Kp + (size_t)(t0 + 16 + lh) * DHc + bSel + 16);
      s1 = wmma16(qf0, kb1, s1);
      v16h kb2 = ld2(Kp + (size_t)(t0 + lh) * DHc + 32 + bSel,
                     Kp + (size_t)(t0 + lh) * DHc + 32 + bSel + 16);
      s0 = wmma16(qf1, kb2, s0);
      v16h kb3 = ld2(Kp + (size_t)(t0 + 16 + lh) * DHc + 32 + bSel,
                     Kp + (size_t)(t0 + 16 + lh) * DHc + 32 + bSel + 16);
      s1 = wmma16(qf1, kb3, s1);
    }
    float p0[8], p1[8], fac[8];
#pragma unroll
    for (int j = 0; j < 8; ++j) {
      float a = s0[j] * tf, c = s1[j] * tf;
      float mx = fmaxf(a, c);
      mx = fmaxf(mx, __shfl_xor(mx, 1));
      mx = fmaxf(mx, __shfl_xor(mx, 2));
      mx = fmaxf(mx, __shfl_xor(mx, 4));
      mx = fmaxf(mx, __shfl_xor(mx, 8));       // row max within 16-lane group
      float mn = fmaxf(mi[j], mx);
      fac[j] = __expf(mi[j] - mn);
      p0[j] = __expf(a - mn);
      p1[j] = __expf(c - mn);
      float rs = p0[j] + p1[j];
      rs += __shfl_xor(rs, 1);
      rs += __shfl_xor(rs, 2);
      rs += __shfl_xor(rs, 4);
      rs += __shfl_xor(rs, 8);
      li[j] = li[j] * fac[j] + rs;
      mi[j] = mn;
    }
#pragma unroll
    for (int nt = 0; nt < 4; ++nt)
#pragma unroll
      for (int j = 0; j < 8; ++j) acc[nt][j] *= fac[j];

    // P: C-layout -> A-layout through LDS
#pragma unroll
    for (int j = 0; j < 8; ++j) {
      int row = (lane < 16) ? j : (j + 8);
      Psh[w][row][lh]      = (_Float16)p0[j];
      Psh[w][row][16 + lh] = (_Float16)p1[j];
    }
    __syncthreads();
    v16h pf = ld2(&Psh[w][lh][aSel], &Psh[w][lh][aSel + 16]);
#pragma unroll
    for (int nt = 0; nt < 4; ++nt) {
      // B frag from V (d-major): col d = nt*16+lh, halves = tokens t0+bSel+h
      v16h vf = ld2(Vp + (size_t)(nt * 16 + lh) * HWc + t0 + bSel,
                    Vp + (size_t)(nt * 16 + lh) * HWc + t0 + bSel + 16);
      acc[nt] = wmma16(pf, vf, acc[nt]);
    }
    __syncthreads();
  }

#pragma unroll
  for (int nt = 0; nt < 4; ++nt)
#pragma unroll
    for (int j = 0; j < 8; ++j) {
      int row = (lane < 16) ? j : (j + 8);
      int tok = q0 + row;
      int dg = h * DHc + nt * 16 + lh;
      obuf[((size_t)b * HWc + tok) * Cc + dg] = (_Float16)(acc[nt][j] / li[j]);
    }
}

// ---------------------------------------------------------------------------
// LayerNorm over C, fused transpose + residual: r[b][c][p] = LN(y)[t][c]+x
// ---------------------------------------------------------------------------
__global__ __launch_bounds__(256) void k_ln(const float* __restrict__ y,
                                            const float* __restrict__ x,
                                            const float* __restrict__ g,
                                            const float* __restrict__ bb,
                                            float* __restrict__ r) {
  __shared__ float red[256];
  int t = blockIdx.x, tid = threadIdx.x;
  int b = t >> 10, p = t & 1023;
  float v0 = y[(size_t)t * Cc + tid];
  float v1 = y[(size_t)t * Cc + 256 + tid];
  red[tid] = v0 + v1;
  __syncthreads();
  for (int off = 128; off > 0; off >>= 1) {
    if (tid < off) red[tid] += red[tid + off];
    __syncthreads();
  }
  float mu = red[0] * (1.0f / Cc);
  __syncthreads();
  float d0 = v0 - mu, d1 = v1 - mu;
  red[tid] = d0 * d0 + d1 * d1;
  __syncthreads();
  for (int off = 128; off > 0; off >>= 1) {
    if (tid < off) red[tid] += red[tid + off];
    __syncthreads();
  }
  float rstd = rsqrtf(red[0] * (1.0f / Cc) + 1e-5f);
  int c0 = tid, c1 = tid + 256;
  size_t o0 = ((size_t)b * Cc + c0) * HWc + p;
  size_t o1 = ((size_t)b * Cc + c1) * HWc + p;
  r[o0] = d0 * rstd * g[c0] + bb[c0] + x[o0];
  r[o1] = d1 * rstd * g[c1] + bb[c1] + x[o1];
}

// depthwise 3x3 SAME + bias; writes token-major f16 for the pw GEMM
__global__ __launch_bounds__(256) void k_dw(const float* __restrict__ r,
                                            const float* __restrict__ w,
                                            const float* __restrict__ bias,
                                            _Float16* __restrict__ z) {
  int idx = blockIdx.x * 256 + threadIdx.x;   // B*C*HW
  int p = idx & 1023, c = (idx >> 10) & 511, b = idx >> 19;
  int yy = p >> 5, xx = p & 31;
  const float* rp = r + ((size_t)b * Cc + c) * HWc;
  const float* wp = w + c * 9;
  float acc = bias[c];
#pragma unroll
  for (int ky = -1; ky <= 1; ++ky)
#pragma unroll
    for (int kx = -1; kx <= 1; ++kx) {
      int y2 = yy + ky, x2 = xx + kx;
      if (y2 >= 0 && y2 < 32 && x2 >= 0 && x2 < 32)
        acc += rp[y2 * 32 + x2] * wp[(ky + 1) * 3 + (kx + 1)];
    }
  z[((size_t)b * HWc + p) * Cc + c] = (_Float16)acc;
}

// ---------------------------------------------------------------------------
extern "C" void kernel_launch(void* const* d_in, const int* in_sizes, int n_in,
                              void* d_out, int out_size, void* d_ws,
                              size_t ws_size, hipStream_t stream) {
  const float* x      = (const float*)d_in[0];
  const float* qkv_w  = (const float*)d_in[1];
  const float* proj_w = (const float*)d_in[2];
  const float* proj_b = (const float*)d_in[3];
  const float* temp   = (const float*)d_in[4];
  const float* ln_g   = (const float*)d_in[5];
  const float* ln_b   = (const float*)d_in[6];
  const float* pos    = (const float*)d_in[7];
  const float* dw_w   = (const float*)d_in[8];
  const float* dw_b   = (const float*)d_in[9];
  const float* pw_w   = (const float*)d_in[10];
  const float* pw_b   = (const float*)d_in[11];
  float* out = (float*)d_out;

  char* ws = (char*)d_ws;
  size_t off = 0;
  auto alloc = [&](size_t bytes) -> void* {
    void* p = ws + off;
    off = (off + bytes + 255) & ~(size_t)255;
    return p;
  };
  _Float16* tkn   = (_Float16*)alloc((size_t)Mc * Cc * 2);
  _Float16* qkvW  = (_Float16*)alloc((size_t)Cc * 3 * Cc * 2);
  _Float16* projW = (_Float16*)alloc((size_t)Cc * Cc * 2);
  _Float16* pwW   = (_Float16*)alloc((size_t)Cc * Cc * 2);
  _Float16* qb    = (_Float16*)alloc((size_t)Mc * Cc * 2);
  _Float16* kb    = (_Float16*)alloc((size_t)Mc * Cc * 2);
  _Float16* vb    = (_Float16*)alloc((size_t)Mc * Cc * 2);
  _Float16* obuf  = (_Float16*)alloc((size_t)Mc * Cc * 2);
  float*    ybuf  = (float*)alloc((size_t)Mc * Cc * 4);
  float*    rbuf  = (float*)alloc((size_t)Mc * Cc * 4);
  _Float16* zbuf  = (_Float16*)alloc((size_t)Mc * Cc * 2);

  // 1) weight convert/transpose -> f16 [K][N]
  k_wt<<<(3 * Cc * Cc) / 256, 256, 0, stream>>>(qkv_w, qkvW, 3 * Cc, Cc);
  k_wt<<<(Cc * Cc) / 256, 256, 0, stream>>>(proj_w, projW, Cc, Cc);
  k_wt<<<(Cc * Cc) / 256, 256, 0, stream>>>(pw_w, pwW, Cc, Cc);
  // 2) tokens
  k_tokens<<<(Mc * Cc) / 256, 256, 0, stream>>>(x, pos, tkn);
  // 3) qkv GEMM (M=8192, N=1536, K=512)
  k_gemm<0><<<dim3(3 * Cc / 64, Mc / 64), 128, 0, stream>>>(
      tkn, qkvW, 3 * Cc, Cc, nullptr, qb, kb, vb, nullptr, nullptr, nullptr);
  // 4) token-axis L2 norm of q and k
  k_normqk<<<Bc * NHc * 2, 256, 0, stream>>>(qb, kb);
  // 5) attention
  k_attn<<<Bc * NHc * (HWc / 64), 128, 0, stream>>>(qb, kb, vb, temp, obuf);
  // 6) proj GEMM (+bias) -> f32 y
  k_gemm<1><<<dim3(Cc / 64, Mc / 64), 128, 0, stream>>>(
      obuf, projW, Cc, Cc, proj_b, nullptr, nullptr, nullptr, ybuf, nullptr,
      nullptr);
  // 7) LN + transpose + residual -> r
  k_ln<<<Mc, 256, 0, stream>>>(ybuf, x, ln_g, ln_b, rbuf);
  // 8) depthwise 3x3 -> z (token-major f16)
  k_dw<<<(Mc * Cc) / 256, 256, 0, stream>>>(rbuf, dw_w, dw_b, zbuf);
  // 9) pointwise GEMM (+bias) + residual -> out
  k_gemm<2><<<dim3(Cc / 64, Mc / 64), 128, 0, stream>>>(
      zbuf, pwW, Cc, Cc, pw_b, nullptr, nullptr, nullptr, nullptr, rbuf, out);
}
